// MOETransformerLayer_22127671509754
// MI455X (gfx1250) — matrix operand order repaired and checked by reading, not compiled, not run on hardware
//
#include <hip/hip_runtime.h>
#include <cstdint>

typedef __attribute__((ext_vector_type(16))) __bf16 v16bf;
typedef __attribute__((ext_vector_type(8)))  float  v8f;
typedef __attribute__((ext_vector_type(4)))  unsigned int v4u;
typedef __attribute__((ext_vector_type(4)))  int v4i;
typedef __attribute__((ext_vector_type(8)))  int v8i;
using u16t = unsigned short;

constexpr int cB = 2, cS = 2048, cM = 1024, cH = 16, cDH = 64, cE = 8, cDFF = 4096;
constexpr int cT = cB * cS;

#if __has_builtin(__builtin_amdgcn_tensor_load_to_lds) && __has_builtin(__builtin_amdgcn_s_wait_tensorcnt)
#define USE_TDM 1
#else
#define USE_TDM 0
#endif

__device__ inline u16t f2bf(float f) {
  union { float f; unsigned u; } x; x.f = f;
  unsigned r = x.u + 0x7FFFu + ((x.u >> 16) & 1u);
  return (u16t)(r >> 16);
}
__device__ inline float bf2f(u16t h) {
  union { unsigned u; float f; } x; x.u = ((unsigned)h) << 16;
  return x.f;
}

union FragBF { v16bf v; uint4 q[2]; u16t s[16]; };

// A fragment (16x32, rows=M, K=32): lane<16 holds row=lane, K=0..15;
// lane>=16 holds row=lane-16, K=16..31.
__device__ inline v16bf load_a_frag(const u16t* tile, int lda, int rowbase) {
  int lane = threadIdx.x & 31;
  int r = lane & 15, hi = lane >> 4;
  const u16t* p = tile + (size_t)(rowbase + r) * lda + hi * 16;
  FragBF u;
  u.q[0] = *(const uint4*)(p);
  u.q[1] = *(const uint4*)(p + 8);
  return u.v;
}
// B fragment (32x16, K x N) from an N-major tile (tile[n][k]):
// lane<16: col n=lane, K=0..15 contiguous; lane>=16: col n=lane-16, K=16..31.
__device__ inline v16bf load_b_frag(const u16t* tile, int lda, int colbase) {
  int lane = threadIdx.x & 31;
  int n = lane & 15, hi = lane >> 4;
  const u16t* p = tile + (size_t)(colbase + n) * lda + hi * 16;
  FragBF u;
  u.q[0] = *(const uint4*)(p);
  u.q[1] = *(const uint4*)(p + 8);
  return u.v;
}
__device__ inline v8f wmma_bf16(v16bf a, v16bf b, v8f c) {
  return __builtin_amdgcn_wmma_f32_16x16x32_bf16(false, a, false, b, (short)0, c, false, false);
}

#if USE_TDM
// TDM: async copy of a 128-row x 32-col bf16 tile (row stride = k_elems) from
// global into LDS.  pad_enable: every 16 DWORDs (=32 bf16) insert 4 DWORDs of
// padding -> LDS row stride 40 u16, matching load_a_frag's lda=40 layout.
// D# layout per CDNA5 ISA 8.3/8.4 (group0: count/lds/global/type, group1:
// data_size=2B, pad_interval=3(16DW), pad_amount=3(4DW), dims/strides).
__device__ inline void tdm_load_tile_bf16(const void* gsrc, void* lds_dst,
                                          unsigned k_elems, unsigned rows_total) {
  unsigned long long ga = (unsigned long long)(uintptr_t)gsrc;
  unsigned lo = (unsigned)(unsigned long long)(uintptr_t)lds_dst;
  v4u g0;
  g0[0] = 1u;                                     // count=1, user descriptor
  g0[1] = lo;                                     // lds_addr (bytes)
  g0[2] = (unsigned)ga;                           // global_addr[31:0]
  g0[3] = (unsigned)((ga >> 32) & 0x01FFFFFFull)  // global_addr[56:32]
          | (2u << 30);                           // type = 2 ("image")
  unsigned d0 = (1u << 16)     // data_size = 1 -> 2 bytes
              | (1u << 20)     // pad_enable
              | (3u << 22)     // pad_interval: 16 DWORDs
              | (3u << 25);    // pad_amount:   4 DWORDs
  v8i g1;
  g1[0] = (int)d0;
  g1[1] = (int)((k_elems & 0xFFFFu) << 16);                       // tensor_dim0 lo
  g1[2] = (int)((k_elems >> 16) | ((rows_total & 0xFFFFu) << 16));// dim0 hi | dim1 lo
  g1[3] = (int)((rows_total >> 16) | (32u << 16));                // dim1 hi | tile_dim0=32
  g1[4] = 128;                                                    // tile_dim1=128, tile_dim2=0
  g1[5] = (int)k_elems;                                           // tensor_dim0_stride lo
  g1[6] = 0;                                                      // stride0 hi | stride1 lo
  g1[7] = 0;
  v4i z4 = {0, 0, 0, 0};
#if defined(__clang_major__) && __clang_major__ >= 23
  v8i z8 = {0, 0, 0, 0, 0, 0, 0, 0};
  __builtin_amdgcn_tensor_load_to_lds(g0, g1, z4, z4, z8, 0);
#else
  __builtin_amdgcn_tensor_load_to_lds(g0, g1, z4, z4, 0);
#endif
}
__device__ inline void tdm_wait() { __builtin_amdgcn_s_wait_tensorcnt(0); }
#endif

// ---------------------------------------------------------------- utilities
__global__ void zero_kernel(float* acc, int* count, long n) {
  long i = (long)blockIdx.x * blockDim.x + threadIdx.x;
  if (i < n) acc[i] = 0.f;
  if (i < cE) count[i] = 0;
}

__global__ __launch_bounds__(256) void ln_bf16_kernel(const float* __restrict__ x,
                                                      const float* __restrict__ g,
                                                      const float* __restrict__ b,
                                                      u16t* __restrict__ out) {
  int t = blockIdx.x;
  const float* xr = x + (size_t)t * cM;
  float s = 0.f, s2 = 0.f;
  for (int i = threadIdx.x; i < cM; i += 256) { float v = xr[i]; s += v; s2 += v * v; }
  for (int m = 1; m < 32; m <<= 1) { s += __shfl_xor(s, m, 32); s2 += __shfl_xor(s2, m, 32); }
  __shared__ float rs[8], rs2[8];
  int wid = threadIdx.x >> 5;
  if ((threadIdx.x & 31) == 0) { rs[wid] = s; rs2[wid] = s2; }
  __syncthreads();
  float S = 0.f, S2 = 0.f;
  for (int w = 0; w < 8; w++) { S += rs[w]; S2 += rs2[w]; }
  float mu = S / cM;
  float var = S2 / cM - mu * mu;
  float rinv = rsqrtf(var + 1e-5f);
  for (int i = threadIdx.x; i < cM; i += 256)
    out[(size_t)t * cM + i] = f2bf((xr[i] - mu) * rinv * g[i] + b[i]);
}

// -------------------------------------------------- QKV projection (WMMA)
__global__ __launch_bounds__(256) void gemm_qkv_kernel(const u16t* __restrict__ A,
                                                       const float* __restrict__ W,
                                                       const float* __restrict__ bias,
                                                       u16t* __restrict__ qo,
                                                       u16t* __restrict__ ko,
                                                       u16t* __restrict__ vo) {
  constexpr int K = cM;
  __shared__ __align__(16) u16t As[128 * 40];
  __shared__ __align__(16) u16t Ws[64 * 40];
  int tn = blockIdx.x * 64, tm = blockIdx.y * 128;
  int tid = threadIdx.x, lane = tid & 31, wid = tid >> 5;
  int wm = (wid & 3) * 32, wn = (wid >> 2) * 32;
  v8f z = {0, 0, 0, 0, 0, 0, 0, 0};
  v8f c[2][2] = {{z, z}, {z, z}};
  int wrow = tid >> 2, wq = tid & 3;
#if !USE_TDM
  int arow = tid >> 1, ahalf = tid & 1;
#endif
  for (int k0 = 0; k0 < K; k0 += 32) {
    __syncthreads();
#if USE_TDM
    if (wid == 0)
      tdm_load_tile_bf16(A + (size_t)tm * K + k0, As, (unsigned)K, (unsigned)cT);
#else
    { const uint4* src = (const uint4*)(A + (size_t)(tm + arow) * K + k0 + ahalf * 16);
      uint4* dst = (uint4*)(As + arow * 40 + ahalf * 16);
      dst[0] = src[0]; dst[1] = src[1]; }
#endif
    { const float* wp = W + (size_t)(tn + wrow) * K + k0 + wq * 8;
      if (k0 + 32 < K) __builtin_prefetch((const void*)(wp + 32), 0, 1);
      const float4* src = (const float4*)wp;
      float4 a0 = src[0], a1 = src[1];
      union { u16t s[8]; uint4 q; } cv;
      cv.s[0] = f2bf(a0.x); cv.s[1] = f2bf(a0.y); cv.s[2] = f2bf(a0.z); cv.s[3] = f2bf(a0.w);
      cv.s[4] = f2bf(a1.x); cv.s[5] = f2bf(a1.y); cv.s[6] = f2bf(a1.z); cv.s[7] = f2bf(a1.w);
      *(uint4*)(Ws + wrow * 40 + wq * 8) = cv.q; }
#if USE_TDM
    if (wid == 0) tdm_wait();
#endif
    __syncthreads();
    v16bf aF[2], bF[2];
    aF[0] = load_a_frag(As, 40, wm);
    aF[1] = load_a_frag(As, 40, wm + 16);
    bF[0] = load_b_frag(Ws, 40, wn);
    bF[1] = load_b_frag(Ws, 40, wn + 16);
    for (int sm = 0; sm < 2; sm++)
      for (int sn = 0; sn < 2; sn++)
        c[sm][sn] = wmma_bf16(aF[sm], bF[sn], c[sm][sn]);
  }
  int hi = lane >> 4, nn = lane & 15;
  for (int sm = 0; sm < 2; sm++)
    for (int sn = 0; sn < 2; sn++)
      for (int r = 0; r < 8; r++) {
        int t = tm + wm + sm * 16 + r + hi * 8;
        int n = tn + wn + sn * 16 + nn;
        float val = c[sm][sn][r] + bias[n];
        int cidx = n >> 10, hh = (n >> 6) & 15, d = n & 63;
        int bb = t >> 11, ss = t & (cS - 1);
        size_t dst = (((size_t)bb * cH + hh) * cS + ss) * cDH + d;
        u16t* o = cidx == 0 ? qo : (cidx == 1 ? ko : vo);
        o[dst] = f2bf(val);
      }
}

// -------------------------------------------------- flash attention (WMMA)
__global__ __launch_bounds__(128) void attn_kernel(const u16t* __restrict__ Q,
                                                   const u16t* __restrict__ Kmat,
                                                   const u16t* __restrict__ Vmat,
                                                   u16t* __restrict__ ctx) {
  __shared__ __align__(16) u16t Vs[4][32 * 72];
  __shared__ __align__(16) u16t Ps[4][16 * 40];
  int bh = blockIdx.y;
  int wid = threadIdx.x >> 5, lane = threadIdx.x & 31;
  int qtile = blockIdx.x * 64 + wid * 16;
  int hi = lane >> 4, nn = lane & 15;
  const u16t* qbase = Q + (size_t)bh * cS * cDH;
  const u16t* kbase = Kmat + (size_t)bh * cS * cDH;
  const u16t* vbase = Vmat + (size_t)bh * cS * cDH;
  v16bf qf[2];
  for (int dh = 0; dh < 2; dh++) {
    const u16t* p = qbase + (size_t)(qtile + nn) * cDH + dh * 32 + hi * 16;
    FragBF u; u.q[0] = *(const uint4*)p; u.q[1] = *(const uint4*)(p + 8);
    qf[dh] = u.v;
  }
  v8f z = {0, 0, 0, 0, 0, 0, 0, 0};
  v8f acc[4] = {z, z, z, z};
  float mrun[8], lrun[8];
  for (int r = 0; r < 8; r++) { mrun[r] = -1e30f; lrun[r] = 0.f; }
  int kend = qtile + 15;
  for (int kb = 0; kb <= kend; kb += 32) {
    if (kb + 32 <= kend) {
      __builtin_prefetch((const void*)(kbase + (size_t)(kb + 32 + lane) * cDH), 0, 1);
      __builtin_prefetch((const void*)(vbase + (size_t)(kb + 32 + lane) * cDH), 0, 1);
    }
    // stage V[kb..kb+31][0..63] into LDS (row per lane, b128 stores)
    { const uint4* src = (const uint4*)(vbase + (size_t)(kb + lane) * cDH);
      uint4* dst = (uint4*)(&Vs[wid][lane * 72]);
      for (int i = 0; i < 8; i++) dst[i] = src[i]; }
    // scores 16q x 32k:  S = Q (16x64) @ K^T (64x32)
    v8f s[2] = {z, z};
    for (int f = 0; f < 2; f++)
      for (int dh = 0; dh < 2; dh++) {
        const u16t* p = kbase + (size_t)(kb + f * 16 + nn) * cDH + dh * 32 + hi * 16;
        FragBF u; u.q[0] = *(const uint4*)p; u.q[1] = *(const uint4*)(p + 8);
        s[f] = wmma_bf16(qf[dh], u.v, s[f]);
      }
    int qrow0 = qtile + hi * 8;
    for (int r = 0; r < 8; r++) {
      for (int f = 0; f < 2; f++) {
        int key = kb + f * 16 + nn;
        if (key > qrow0 + r) s[f][r] = -1e30f;
      }
      float bm = fmaxf(s[0][r], s[1][r]);
      for (int m = 1; m < 16; m <<= 1) bm = fmaxf(bm, __shfl_xor(bm, m, 32));
      float mnew = fmaxf(mrun[r], bm);
      float scl = __expf(mrun[r] - mnew);
      mrun[r] = mnew;
      float p0 = __expf(s[0][r] - mnew);
      float p1 = __expf(s[1][r] - mnew);
      Ps[wid][(r + hi * 8) * 40 + nn] = f2bf(p0);
      Ps[wid][(r + hi * 8) * 40 + 16 + nn] = f2bf(p1);
      float rsum = p0 + p1;
      for (int m = 1; m < 16; m <<= 1) rsum += __shfl_xor(rsum, m, 32);
      lrun[r] = lrun[r] * scl + rsum;
      for (int dg = 0; dg < 4; dg++) acc[dg][r] *= scl;
    }
    // ctx += P (16x32) @ V (32x64); per-wave LDS ops are in-order (DScnt)
    v16bf pf = load_a_frag(Ps[wid], 40, 0);
    for (int dg = 0; dg < 4; dg++) {
      FragBF u;
      int d = dg * 16 + nn;
      for (int i = 0; i < 16; i++) u.s[i] = Vs[wid][(hi * 16 + i) * 72 + d];
      acc[dg] = wmma_bf16(pf, u.v, acc[dg]);
    }
  }
  int b = bh >> 4, h = bh & 15;
  for (int dg = 0; dg < 4; dg++)
    for (int r = 0; r < 8; r++) {
      int srow = qtile + r + hi * 8;
      int t = b * cS + srow;
      float val = acc[dg][r] / lrun[r];
      ctx[(size_t)t * cM + h * cDH + dg * 16 + nn] = f2bf(val);
    }
}

// -------------------------------------------------- output proj + residual
__global__ __launch_bounds__(256) void gemm_ow_kernel(const u16t* __restrict__ A,
                                                      const float* __restrict__ W,
                                                      const float* __restrict__ bias,
                                                      const float* __restrict__ xin,
                                                      float* __restrict__ y) {
  constexpr int K = cM;
  __shared__ __align__(16) u16t As[128 * 40];
  __shared__ __align__(16) u16t Ws[64 * 40];
  int tn = blockIdx.x * 64, tm = blockIdx.y * 128;
  int tid = threadIdx.x, lane = tid & 31, wid = tid >> 5;
  int wm = (wid & 3) * 32, wn = (wid >> 2) * 32;
  v8f z = {0, 0, 0, 0, 0, 0, 0, 0};
  v8f c[2][2] = {{z, z}, {z, z}};
  int wrow = tid >> 2, wq = tid & 3;
#if !USE_TDM
  int arow = tid >> 1, ahalf = tid & 1;
#endif
  for (int k0 = 0; k0 < K; k0 += 32) {
    __syncthreads();
#if USE_TDM
    if (wid == 0)
      tdm_load_tile_bf16(A + (size_t)tm * K + k0, As, (unsigned)K, (unsigned)cT);
#else
    { const uint4* src = (const uint4*)(A + (size_t)(tm + arow) * K + k0 + ahalf * 16);
      uint4* dst = (uint4*)(As + arow * 40 + ahalf * 16);
      dst[0] = src[0]; dst[1] = src[1]; }
#endif
    { const float* wp = W + (size_t)(tn + wrow) * K + k0 + wq * 8;
      if (k0 + 32 < K) __builtin_prefetch((const void*)(wp + 32), 0, 1);
      const float4* src = (const float4*)wp;
      float4 a0 = src[0], a1 = src[1];
      union { u16t s[8]; uint4 q; } cv;
      cv.s[0] = f2bf(a0.x); cv.s[1] = f2bf(a0.y); cv.s[2] = f2bf(a0.z); cv.s[3] = f2bf(a0.w);
      cv.s[4] = f2bf(a1.x); cv.s[5] = f2bf(a1.y); cv.s[6] = f2bf(a1.z); cv.s[7] = f2bf(a1.w);
      *(uint4*)(Ws + wrow * 40 + wq * 8) = cv.q; }
#if USE_TDM
    if (wid == 0) tdm_wait();
#endif
    __syncthreads();
    v16bf aF[2], bF[2];
    aF[0] = load_a_frag(As, 40, wm);
    aF[1] = load_a_frag(As, 40, wm + 16);
    bF[0] = load_b_frag(Ws, 40, wn);
    bF[1] = load_b_frag(Ws, 40, wn + 16);
    for (int sm = 0; sm < 2; sm++)
      for (int sn = 0; sn < 2; sn++)
        c[sm][sn] = wmma_bf16(aF[sm], bF[sn], c[sm][sn]);
  }
  int hi = lane >> 4, nn = lane & 15;
  for (int sm = 0; sm < 2; sm++)
    for (int sn = 0; sn < 2; sn++)
      for (int r = 0; r < 8; r++) {
        int t = tm + wm + sm * 16 + r + hi * 8;
        int n = tn + wn + sn * 16 + nn;
        float val = c[sm][sn][r] + bias[n];
        y[(size_t)t * cM + n] = xin[(size_t)t * cM + n] + val;
      }
}

// -------------------------------------------------- MoE gating + routing
__global__ __launch_bounds__(256) void gate_kernel(const u16t* __restrict__ h2,
                                                   const float* __restrict__ gw,
                                                   int* __restrict__ topi,
                                                   float* __restrict__ topw) {
  int wid = threadIdx.x >> 5, lane = threadIdx.x & 31;
  int t = blockIdx.x * 8 + wid;
  float acc[cE];
  for (int e = 0; e < cE; e++) acc[e] = 0.f;
  const u16t* hr = h2 + (size_t)t * cM;
  for (int k = lane; k < cM; k += 32) {
    float hv = bf2f(hr[k]);
    for (int e = 0; e < cE; e++) acc[e] += hv * gw[e * cM + k];
  }
  for (int e = 0; e < cE; e++)
    for (int m = 1; m < 32; m <<= 1) acc[e] += __shfl_xor(acc[e], m, 32);
  if (lane == 0) {
    float mx = acc[0];
    for (int e = 1; e < cE; e++) mx = fmaxf(mx, acc[e]);
    float ex[cE];
    for (int e = 0; e < cE; e++) ex[e] = __expf(acc[e] - mx);
    int i0 = 0;
    for (int e = 1; e < cE; e++) if (acc[e] > acc[i0]) i0 = e;
    int i1 = (i0 == 0) ? 1 : 0;
    for (int e = 0; e < cE; e++) if (e != i0 && acc[e] > acc[i1]) i1 = e;
    float w0 = ex[i0], w1 = ex[i1], sum = w0 + w1;
    topi[t * 2] = i0; topi[t * 2 + 1] = i1;
    topw[t * 2] = w0 / sum; topw[t * 2 + 1] = w1 / sum;
  }
}

__global__ void route_kernel(const int* __restrict__ topi, const float* __restrict__ topw,
                             int* count, int* list, float* wlist) {
  int t = blockIdx.x * blockDim.x + threadIdx.x;
  if (t >= cT) return;
  for (int j = 0; j < 2; j++) {
    int e = topi[t * 2 + j];
    int pos = atomicAdd(&count[e], 1);
    list[e * cT + pos] = t;
    wlist[e * cT + pos] = topw[t * 2 + j];
  }
}

__global__ void offsets_kernel(const int* count, int* offs) {
  if (threadIdx.x == 0 && blockIdx.x == 0) {
    int off = 0;
    for (int e = 0; e < cE; e++) { offs[e] = off; off += (count[e] + 127) & ~127; }
  }
}

// -------------------------------------------------- MoE GEMM1: silu(h@w1^T)*(h@w3^T)
__global__ __launch_bounds__(256) void moe_gemm1_kernel(const u16t* __restrict__ h2,
                                                        const float* __restrict__ w1,
                                                        const float* __restrict__ w3,
                                                        const int* __restrict__ list,
                                                        const int* __restrict__ count,
                                                        const int* __restrict__ offs,
                                                        u16t* __restrict__ interm) {
  constexpr int K = cM;
  int e = blockIdx.z;
  int cnt = count[e];
  int cntR = (cnt + 127) & ~127;
  int tt = blockIdx.y * 128;
  if (tt >= cntR) return;
  __shared__ __align__(16) u16t As[128 * 40];
  __shared__ __align__(16) u16t W1s[64 * 40];
  __shared__ __align__(16) u16t W3s[64 * 40];
  int tn = blockIdx.x * 64;
  int tid = threadIdx.x, lane = tid & 31, wid = tid >> 5;
  int wm = (wid & 3) * 32, wn = (wid >> 2) * 32;
  v8f z = {0, 0, 0, 0, 0, 0, 0, 0};
  v8f c1[2][2] = {{z, z}, {z, z}};
  v8f c3[2][2] = {{z, z}, {z, z}};
  int arow = tid >> 1, ahalf = tid & 1;
  int gidx = tt + arow;
  int token = list[e * cT + ((gidx < cnt) ? gidx : (cnt - 1))];
  const u16t* Arow = h2 + (size_t)token * K;
  int wrow = tid >> 2, wq = tid & 3;
  const float* W1r = w1 + ((size_t)e * cDFF + tn + wrow) * K;
  const float* W3r = w3 + ((size_t)e * cDFF + tn + wrow) * K;
  for (int k0 = 0; k0 < K; k0 += 32) {
    __syncthreads();
    { const uint4* src = (const uint4*)(Arow + k0 + ahalf * 16);
      uint4* dst = (uint4*)(As + arow * 40 + ahalf * 16);
      dst[0] = src[0]; dst[1] = src[1]; }
    { const float* wp = W1r + k0 + wq * 8;
      if (k0 + 32 < K) __builtin_prefetch((const void*)(wp + 32), 0, 1);
      const float4* src = (const float4*)wp;
      float4 a0 = src[0], a1 = src[1];
      union { u16t s[8]; uint4 q; } cv;
      cv.s[0] = f2bf(a0.x); cv.s[1] = f2bf(a0.y); cv.s[2] = f2bf(a0.z); cv.s[3] = f2bf(a0.w);
      cv.s[4] = f2bf(a1.x); cv.s[5] = f2bf(a1.y); cv.s[6] = f2bf(a1.z); cv.s[7] = f2bf(a1.w);
      *(uint4*)(W1s + wrow * 40 + wq * 8) = cv.q; }
    { const float* wp = W3r + k0 + wq * 8;
      if (k0 + 32 < K) __builtin_prefetch((const void*)(wp + 32), 0, 1);
      const float4* src = (const float4*)wp;
      float4 a0 = src[0], a1 = src[1];
      union { u16t s[8]; uint4 q; } cv;
      cv.s[0] = f2bf(a0.x); cv.s[1] = f2bf(a0.y); cv.s[2] = f2bf(a0.z); cv.s[3] = f2bf(a0.w);
      cv.s[4] = f2bf(a1.x); cv.s[5] = f2bf(a1.y); cv.s[6] = f2bf(a1.z); cv.s[7] = f2bf(a1.w);
      *(uint4*)(W3s + wrow * 40 + wq * 8) = cv.q; }
    __syncthreads();
    v16bf aF[2], b1F[2], b3F[2];
    aF[0] = load_a_frag(As, 40, wm);
    aF[1] = load_a_frag(As, 40, wm + 16);
    b1F[0] = load_b_frag(W1s, 40, wn);
    b1F[1] = load_b_frag(W1s, 40, wn + 16);
    b3F[0] = load_b_frag(W3s, 40, wn);
    b3F[1] = load_b_frag(W3s, 40, wn + 16);
    for (int sm = 0; sm < 2; sm++)
      for (int sn = 0; sn < 2; sn++) {
        c1[sm][sn] = wmma_bf16(aF[sm], b1F[sn], c1[sm][sn]);
        c3[sm][sn] = wmma_bf16(aF[sm], b3F[sn], c3[sm][sn]);
      }
  }
  int hi = lane >> 4, nn = lane & 15;
  for (int sm = 0; sm < 2; sm++)
    for (int sn = 0; sn < 2; sn++)
      for (int r = 0; r < 8; r++) {
        float g = c1[sm][sn][r];
        float u = c3[sm][sn][r];
        float fe = (g / (1.f + __expf(-g))) * u;
        int prow = tt + wm + sm * 16 + r + hi * 8;
        size_t slot = (size_t)(offs[e] + prow);
        interm[slot * cDFF + tn + wn + sn * 16 + nn] = f2bf(fe);
      }
}

// -------------------------------------------------- MoE GEMM2 + weighted scatter
__global__ __launch_bounds__(256) void moe_gemm2_kernel(const u16t* __restrict__ interm,
                                                        const float* __restrict__ w2,
                                                        const int* __restrict__ list,
                                                        const float* __restrict__ wlist,
                                                        const int* __restrict__ count,
                                                        const int* __restrict__ offs,
                                                        float* __restrict__ acc) {
  constexpr int K = cDFF;
  int e = blockIdx.z;
  int cnt = count[e];
  int cntR = (cnt + 127) & ~127;
  int tt = blockIdx.y * 128;
  if (tt >= cntR) return;
  __shared__ __align__(16) u16t As[128 * 40];
  __shared__ __align__(16) u16t Ws[64 * 40];
  int tn = blockIdx.x * 64;
  int tid = threadIdx.x, lane = tid & 31, wid = tid >> 5;
  int wm = (wid & 3) * 32, wn = (wid >> 2) * 32;
  v8f z = {0, 0, 0, 0, 0, 0, 0, 0};
  v8f c[2][2] = {{z, z}, {z, z}};
  const u16t* Abase = interm + ((size_t)offs[e] + tt) * K;
  int wrow = tid >> 2, wq = tid & 3;
  const float* Wr = w2 + ((size_t)e * cM + tn + wrow) * K;
#if !USE_TDM
  int arow = tid >> 1, ahalf = tid & 1;
#endif
  for (int k0 = 0; k0 < K; k0 += 32) {
    __syncthreads();
#if USE_TDM
    if (wid == 0)
      tdm_load_tile_bf16(Abase + k0, As, (unsigned)K, (unsigned)(2 * cT + 128 * cE));
#else
    { const uint4* src = (const uint4*)(Abase + (size_t)arow * K + k0 + ahalf * 16);
      uint4* dst = (uint4*)(As + arow * 40 + ahalf * 16);
      dst[0] = src[0]; dst[1] = src[1]; }
#endif
    { const float* wp = Wr + k0 + wq * 8;
      if (k0 + 32 < K) __builtin_prefetch((const void*)(wp + 32), 0, 1);
      const float4* src = (const float4*)wp;
      float4 a0 = src[0], a1 = src[1];
      union { u16t s[8]; uint4 q; } cv;
      cv.s[0] = f2bf(a0.x); cv.s[1] = f2bf(a0.y); cv.s[2] = f2bf(a0.z); cv.s[3] = f2bf(a0.w);
      cv.s[4] = f2bf(a1.x); cv.s[5] = f2bf(a1.y); cv.s[6] = f2bf(a1.z); cv.s[7] = f2bf(a1.w);
      *(uint4*)(Ws + wrow * 40 + wq * 8) = cv.q; }
#if USE_TDM
    if (wid == 0) tdm_wait();
#endif
    __syncthreads();
    v16bf aF[2], bF[2];
    aF[0] = load_a_frag(As, 40, wm);
    aF[1] = load_a_frag(As, 40, wm + 16);
    bF[0] = load_b_frag(Ws, 40, wn);
    bF[1] = load_b_frag(Ws, 40, wn + 16);
    for (int sm = 0; sm < 2; sm++)
      for (int sn = 0; sn < 2; sn++)
        c[sm][sn] = wmma_bf16(aF[sm], bF[sn], c[sm][sn]);
  }
  int hi = lane >> 4, nn = lane & 15;
  for (int sm = 0; sm < 2; sm++)
    for (int sn = 0; sn < 2; sn++)
      for (int r = 0; r < 8; r++) {
        int prow = tt + wm + sm * 16 + r + hi * 8;
        if (prow < cnt) {
          int token = list[e * cT + prow];
          float w = wlist[e * cT + prow];
          int n = tn + wn + sn * 16 + nn;
          atomicAdd(&acc[(size_t)token * cM + n], w * c[sm][sn][r]);
        }
      }
}

__global__ void final_add_kernel(const float* __restrict__ y, const float* __restrict__ acc,
                                 float* __restrict__ out) {
  int i = blockIdx.x * 256 + threadIdx.x;
  out[i] = y[i] + acc[i];
}

// ---------------------------------------------------------------- launcher
extern "C" void kernel_launch(void* const* d_in, const int* in_sizes, int n_in,
                              void* d_out, int out_size, void* d_ws, size_t ws_size,
                              hipStream_t stream) {
  const float* x     = (const float*)d_in[0];
  const float* ln1_g = (const float*)d_in[1];
  const float* ln1_b = (const float*)d_in[2];
  const float* qkvw  = (const float*)d_in[3];
  const float* qkvb  = (const float*)d_in[4];
  const float* ow    = (const float*)d_in[5];
  const float* ob    = (const float*)d_in[6];
  const float* ln2_g = (const float*)d_in[7];
  const float* ln2_b = (const float*)d_in[8];
  const float* gatew = (const float*)d_in[9];
  const float* w1    = (const float*)d_in[10];
  const float* w2    = (const float*)d_in[11];
  const float* w3    = (const float*)d_in[12];
  float* out = (float*)d_out;
  (void)in_sizes; (void)n_in; (void)out_size; (void)ws_size;

  char* ws = (char*)d_ws;
  size_t off = 0;
  auto alloc = [&](size_t bytes) -> void* {
    void* p = ws + off;
    off += (bytes + 255) & ~(size_t)255;
    return p;
  };
  u16t* h1    = (u16t*)alloc((size_t)cT * cM * 2);
  u16t* qb    = (u16t*)alloc((size_t)cT * cM * 2);
  u16t* kb    = (u16t*)alloc((size_t)cT * cM * 2);
  u16t* vb    = (u16t*)alloc((size_t)cT * cM * 2);
  u16t* ctx   = (u16t*)alloc((size_t)cT * cM * 2);
  float* y    = (float*)alloc((size_t)cT * cM * 4);
  u16t* h2    = (u16t*)alloc((size_t)cT * cM * 2);
  int* topi   = (int*)alloc((size_t)cT * 2 * 4);
  float* topw = (float*)alloc((size_t)cT * 2 * 4);
  int* count  = (int*)alloc(cE * 4);
  int* offs   = (int*)alloc(cE * 4);
  int* list   = (int*)alloc((size_t)cE * cT * 4);
  float* wlist= (float*)alloc((size_t)cE * cT * 4);
  u16t* interm= (u16t*)alloc((size_t)(2 * cT + 128 * cE) * cDFF * 2);
  float* accb = (float*)alloc((size_t)cT * cM * 4);

  long total = (long)cT * cM;
  zero_kernel<<<(total + 255) / 256, 256, 0, stream>>>(accb, count, total);
  ln_bf16_kernel<<<cT, 256, 0, stream>>>(x, ln1_g, ln1_b, h1);
  gemm_qkv_kernel<<<dim3(48, 32), 256, 0, stream>>>(h1, qkvw, qkvb, qb, kb, vb);
  attn_kernel<<<dim3(cS / 64, cB * cH), 128, 0, stream>>>(qb, kb, vb, ctx);
  gemm_ow_kernel<<<dim3(16, 32), 256, 0, stream>>>(ctx, ow, ob, x, y);
  ln_bf16_kernel<<<cT, 256, 0, stream>>>(y, ln2_g, ln2_b, h2);
  gate_kernel<<<cT / 8, 256, 0, stream>>>(h2, gatew, topi, topw);
  route_kernel<<<cT / 256, 256, 0, stream>>>(topi, topw, count, list, wlist);
  offsets_kernel<<<1, 1, 0, stream>>>(count, offs);
  moe_gemm1_kernel<<<dim3(cDFF / 64, cT / 128, cE), 256, 0, stream>>>(h2, w1, w3, list, count, offs, interm);
  moe_gemm2_kernel<<<dim3(cM / 64, cT / 128, cE), 256, 0, stream>>>(interm, w2, list, wlist, count, offs, accb);
  final_add_kernel<<<(int)(total / 256), 256, 0, stream>>>(y, accb, out);
}
